// RGNN_3865470567040
// MI455X (gfx1250) — matrix operand order repaired and checked by reading, not compiled, not run on hardware
//
#include <hip/hip_runtime.h>
#include <hip/hip_bf16.h>
#include <stdint.h>

#define NAT 200000
#define EAT 400000
#define NRG 50000
#define ERG 100000
#define NB  10000
#define CD  128
#define C3  384
#define KMAX 160   // max padded K (weights panels are [<=128+pad] wide)

typedef __bf16 bf16;
typedef __attribute__((ext_vector_type(16))) __bf16 v16bf;
typedef __attribute__((ext_vector_type(8)))  float  v8f;

union FragBF { v16bf v; uint32_t w[8]; uint4 q[2]; };

__device__ __forceinline__ float leakyf(float v){ return v > 0.f ? v : 0.01f * v; }
__device__ __forceinline__ float eluf(float v){ return v > 0.f ? v : (__expf(v) - 1.f); }
__device__ __forceinline__ float sigmf(float v){ return 1.f / (1.f + __expf(-v)); }
__device__ __forceinline__ bf16 tobf(float f){
  uint32_t u = __float_as_uint(f);
  u += 0x7fffu + ((u >> 16) & 1u);
  uint16_t h = (uint16_t)(u >> 16);
  return __builtin_bit_cast(__bf16, h);
}
__device__ __forceinline__ void atomicMaxF(float* addr, float val){
  unsigned* ua = (unsigned*)addr;
  unsigned old = *ua;
  while (true) {
    float oldf = __uint_as_float(old);
    if (oldf >= val) break;
    unsigned assumed = old;
    old = atomicCAS(ua, assumed, __float_as_uint(val));
    if (old == assumed) break;
  }
}

// ---------------------------------------------------------------------------
// WMMA GEMM: Cout[M,N] = act( A[M,K](bf16) @ W[N,K](bf16)^T + bias )
// K multiple of 32; M multiple of 16; N multiple of 128.
// Block: 256 threads = 8 waves; wave w owns M-tile (blockIdx.x*8+w) and the
// full 128-column N chunk (8 accumulators). Weight panel [128 x K] staged in
// LDS once per block; A fragment loaded once per k-step (2x b128), reused by
// 8 WMMAs whose B fragments come from LDS as 2x ds_load_b128 each.
// ---------------------------------------------------------------------------
__global__ __launch_bounds__(256)
void k_gemm(const bf16* __restrict__ A, const bf16* __restrict__ B,
            const float* __restrict__ bias, float* __restrict__ Cout,
            int M, int N, int K, int act)
{
  __shared__ bf16 Bs[128 * KMAX];                 // 40 KB
  const int lane = threadIdx.x & 31;
  const int wv   = threadIdx.x >> 5;              // 0..7
  const int mt   = blockIdx.x * 8 + wv;           // M tile of this wave
  const int n0   = blockIdx.y * 128;              // N chunk base

  // cooperative stage of the 128 x K weight panel (rows n0..n0+127 contiguous)
  {
    const uint4* Bg = (const uint4*)(B + (size_t)n0 * K);
    uint4* Bl = (uint4*)Bs;
    const int nchunks = (128 * K) / 8;            // 16B chunks (8 bf16 each)
    for (int i = threadIdx.x; i < nchunks; i += 256) Bl[i] = Bg[i];
  }
  __syncthreads();

  if (mt * 16 >= M) return;                       // wave-uniform tail guard

  const int r15 = lane & 15, hi = lane >> 4;
  const bf16* Ar = A + (size_t)(mt * 16 + r15) * K;
  const int ab = hi * 8, bb = hi * 8;

  v8f acc[8];
#pragma unroll
  for (int t = 0; t < 8; ++t) acc[t] = (v8f){};

  for (int k0 = 0; k0 < K; k0 += 32) {
    FragBF fa;
    fa.q[0] = *(const uint4*)(Ar + k0 + ab);        // k = ab .. ab+7
    fa.q[1] = *(const uint4*)(Ar + k0 + ab + 16);   // k = ab+16 .. ab+23
#pragma unroll
    for (int t = 0; t < 8; ++t) {
      FragBF fb;
      const bf16* Bl = Bs + (size_t)(t * 16 + r15) * K + k0 + bb;
      fb.q[0] = *(const uint4*)(Bl);                // k = bb .. bb+7
      fb.q[1] = *(const uint4*)(Bl + 16);           // k = bb+16 .. bb+23
      acc[t] = __builtin_amdgcn_wmma_f32_16x16x32_bf16(false, fa.v, false, fb.v,
                                                       (short)0, acc[t], false, false);
    }
  }

#pragma unroll
  for (int t = 0; t < 8; ++t) {
    int n = n0 + t * 16 + r15;
    float bval = bias ? bias[n] : 0.f;
#pragma unroll
    for (int r = 0; r < 8; ++r) {
      int m = mt * 16 + r + hi * 8;
      float v = acc[t][r] + bval;
      if (act == 1) v = leakyf(v);
      Cout[(size_t)m * N + n] = v;
    }
  }
}

// ---------------------------------------------------------------------------
// Elementwise / graph kernels
// ---------------------------------------------------------------------------
__global__ void k_fill(float* p, float v, size_t n){
  for (size_t i = (size_t)blockIdx.x * blockDim.x + threadIdx.x; i < n;
       i += (size_t)gridDim.x * blockDim.x) p[i] = v;
}
__global__ void k_copy(const float* s, float* d, size_t n){
  for (size_t i = (size_t)blockIdx.x * blockDim.x + threadIdx.x; i < n;
       i += (size_t)gridDim.x * blockDim.x) d[i] = s[i];
}
// f32 [M,K] -> bf16 [M,Kp] with zero pad
__global__ void k_cvt2d(const float* __restrict__ src, bf16* __restrict__ dst,
                        int M, int K, int Kp){
  size_t tot = (size_t)M * Kp;
  for (size_t i = (size_t)blockIdx.x * blockDim.x + threadIdx.x; i < tot;
       i += (size_t)gridDim.x * blockDim.x) {
    size_t row = i / Kp; int col = (int)(i % Kp);
    float f = (col < K) ? src[row * K + col] : 0.f;
    dst[i] = tobf(f);
  }
}
// dst[e,0:CD]=X[src[e]]; dst[e,CD:CD+ed]=Eat[e]; rest 0  (src==null -> row e)
__global__ void k_gather(const float* __restrict__ X, const float* __restrict__ Eat,
                         int ed, const int* __restrict__ src, bf16* __restrict__ dst,
                         size_t E, int Kp){
  size_t tot = E * Kp;
  for (size_t i = (size_t)blockIdx.x * blockDim.x + threadIdx.x; i < tot;
       i += (size_t)gridDim.x * blockDim.x) {
    size_t e = i / Kp; int c = (int)(i % Kp);
    int r = src ? src[e] : (int)e;
    float f = 0.f;
    if (c < CD)            f = X[(size_t)r * CD + c];
    else if (c < CD + ed)  f = Eat[e * ed + (c - CD)];
    dst[i] = tobf(f);
  }
}
__global__ void k_rowdot(const float* __restrict__ X, const float* __restrict__ w,
                         float* __restrict__ out, int N){
  for (size_t n = (size_t)blockIdx.x * blockDim.x + threadIdx.x; n < (size_t)N;
       n += (size_t)gridDim.x * blockDim.x) {
    const float* r = X + n * CD;
    float acc = 0.f;
#pragma unroll 8
    for (int c = 0; c < CD; ++c) acc += r[c] * w[c];
    out[n] = acc;
  }
}
// a[e] = leaky((aj?aj[e]:0) + (sS?sS[src? src[e]:e]:0) + sD[dst[e]]); max into maxb
__global__ void k_escore(const float* aj, const float* sS, const float* __restrict__ sD,
                         const int* src, const int* __restrict__ dst,
                         float* a, float* maxb, size_t E){
  for (size_t e = (size_t)blockIdx.x * blockDim.x + threadIdx.x; e < E;
       e += (size_t)gridDim.x * blockDim.x) {
    float v = sD[dst[e]];
    if (aj) v += aj[e];
    if (sS) v += sS[src ? src[e] : (int)e];
    v = leakyf(v);
    a[e] = v;
    atomicMaxF(&maxb[dst[e]], v);
  }
}
__global__ void k_eexp(float* a, const float* __restrict__ maxb,
                       const int* __restrict__ dst, float* sumb, size_t E){
  for (size_t e = (size_t)blockIdx.x * blockDim.x + threadIdx.x; e < E;
       e += (size_t)gridDim.x * blockDim.x) {
    float m = maxb[dst[e]];
    if (m < -9e29f) m = 0.f;                  // isfinite() guard
    float ev = __expf(a[e] - m);
    a[e] = ev;
    atomicAdd(&sumb[dst[e]], ev);
  }
}
__global__ void k_enorm(float* a, const float* __restrict__ sumb,
                        const int* __restrict__ dst, size_t E){
  for (size_t e = (size_t)blockIdx.x * blockDim.x + threadIdx.x; e < E;
       e += (size_t)gridDim.x * blockDim.x)
    a[e] = a[e] / (sumb[dst[e]] + 1e-16f);
}
// agg[dst[e],c] += H[row,c] * (alpha?alpha[e]:1); row = src?src[e]:e
__global__ void k_scatter(const float* __restrict__ H, const int* src,
                          const int* __restrict__ dst, const float* alpha,
                          float* agg, size_t E){
  size_t tot = E * CD;
  for (size_t i = (size_t)blockIdx.x * blockDim.x + threadIdx.x; i < tot;
       i += (size_t)gridDim.x * blockDim.x) {
    size_t e = i / CD; int c = (int)(i % CD);
    int r = src ? src[e] : (int)e;
    float al = alpha ? alpha[e] : 1.f;
    atomicAdd(&agg[(size_t)dst[e] * CD + c], H[(size_t)r * CD + c] * al);
  }
}
// out = relu?( (elu?( agg+bias )) + t + res )
__global__ void k_finalize(const float* __restrict__ agg, const float* bias,
                           const float* tadd, const float* radd, float* out,
                           int N, int do_elu, int do_relu){
  size_t tot = (size_t)N * CD;
  for (size_t i = (size_t)blockIdx.x * blockDim.x + threadIdx.x; i < tot;
       i += (size_t)gridDim.x * blockDim.x) {
    int c = (int)(i % CD);
    float v = agg[i] + (bias ? bias[c] : 0.f);
    if (do_elu) v = eluf(v);
    if (tadd) v += tadd[i];
    if (radd) v += radd[i];
    if (do_relu) v = fmaxf(v, 0.f);
    out[i] = v;
  }
}
__global__ void k_count(const int* __restrict__ tid, float* cnt, int n){
  for (size_t i = (size_t)blockIdx.x * blockDim.x + threadIdx.x; i < (size_t)n;
       i += (size_t)gridDim.x * blockDim.x) atomicAdd(&cnt[tid[i]], 1.f);
}
__global__ void k_div(float* base, const float* __restrict__ cnt, int N){
  size_t tot = (size_t)N * CD;
  for (size_t i = (size_t)blockIdx.x * blockDim.x + threadIdx.x; i < tot;
       i += (size_t)gridDim.x * blockDim.x)
    base[i] = base[i] / fmaxf(cnt[i / CD], 1.f);
}
__global__ void k_gru(const float* __restrict__ gi, const float* __restrict__ gh,
                      const float* __restrict__ hprev, float* out, int N, int do_relu){
  size_t tot = (size_t)N * CD;
  for (size_t i = (size_t)blockIdx.x * blockDim.x + threadIdx.x; i < tot;
       i += (size_t)gridDim.x * blockDim.x) {
    size_t n = i / CD; int c = (int)(i % CD);
    const float* gin = gi + n * C3;
    const float* ghn = gh + n * C3;
    float r  = sigmf(gin[c] + ghn[c]);
    float z  = sigmf(gin[CD + c] + ghn[CD + c]);
    float nn = tanhf(gin[2 * CD + c] + r * ghn[2 * CD + c]);
    float o  = (1.f - z) * nn + z * hprev[i];
    if (do_relu) o = fmaxf(o, 0.f);
    out[i] = o;
  }
}
__global__ void k_pred(const float* __restrict__ fp, const float* __restrict__ W,
                       const float* __restrict__ b, float* pred, int Nb){
  for (size_t n = (size_t)blockIdx.x * blockDim.x + threadIdx.x; n < (size_t)Nb;
       n += (size_t)gridDim.x * blockDim.x) {
    const float* r = fp + n * CD;
    float acc = 0.f;
#pragma unroll 8
    for (int c = 0; c < CD; ++c) acc += r[c] * W[c];
    pred[n] = acc + b[0];
  }
}

// ---------------------------------------------------------------------------
static inline int nblk(size_t n){
  size_t b = (n + 255) / 256;
  return (int)(b > 1048576 ? 1048576 : (b ? b : 1));
}

extern "C" void kernel_launch(void* const* d_in, const int* in_sizes, int n_in,
                              void* d_out, int out_size, void* d_ws, size_t ws_size,
                              hipStream_t stream) {
  (void)in_sizes; (void)n_in; (void)out_size; (void)ws_size;
#define FP(i) ((const float*)d_in[(i)])
  const float* x_in    = FP(0);
  const float* eattr   = FP(1);
  const float* rg_x    = FP(2);
  const float* rg_ea   = FP(3);
  const int*  eidx     = (const int*)d_in[4];
  const int*  pool_sid = (const int*)d_in[5];
  const int*  pool_tid = (const int*)d_in[6];
  const int*  reidx    = (const int*)d_in[7];
  const int*  rg_batch = (const int*)d_in[8];
  const int* srcA = eidx,  *dstA = eidx + EAT;
  const int* srcR = reidx, *dstR = reidx + ERG;

  // ---- workspace suballocation -------------------------------------------
  char* wsp = (char*)d_ws;
  auto alloc = [&](size_t bytes)->void*{
    void* p = wsp; wsp += (bytes + 255) & ~(size_t)255; return p;
  };
  float* x1      = (float*)alloc((size_t)NAT * CD * 4);   // atom features (x1/x2/x3)
  float* res0    = (float*)alloc((size_t)NAT * CD * 4);
  float* tA      = (float*)alloc((size_t)NAT * CD * 4);   // leaky(alin(x))
  float* mA      = (float*)alloc((size_t)NAT * CD * 4);   // messages / hs
  float* hE      = (float*)alloc((size_t)EAT * CD * 4);   // edge hidden; first NAT*CD reused as agg
  float* aggA    = hE;
  float* rg_base = (float*)alloc((size_t)NRG * CD * 4);
  float* rg_out  = (float*)alloc((size_t)NRG * CD * 4);
  float* hR      = (float*)alloc((size_t)NRG * CD * 4);
  float* tR      = (float*)alloc((size_t)NRG * CD * 4);
  float* aggR    = (float*)alloc((size_t)NRG * CD * 4);
  float* giR     = (float*)alloc((size_t)NRG * C3 * 4);
  float* ghR     = (float*)alloc((size_t)NRG * C3 * 4);
  float* hER     = (float*)alloc((size_t)ERG * CD * 4);
  float* outM    = (float*)alloc((size_t)NB * CD * 4);
  float* hB      = (float*)alloc((size_t)NB * CD * 4);
  float* aggB    = (float*)alloc((size_t)NB * CD * 4);
  float* giB     = (float*)alloc((size_t)NB * C3 * 4);
  float* ghB     = (float*)alloc((size_t)NB * C3 * 4);
  float* aj      = (float*)alloc((size_t)EAT * 4);
  float* alpha   = (float*)alloc((size_t)EAT * 4);
  float* sS      = (float*)alloc((size_t)NAT * 4);
  float* sD      = (float*)alloc((size_t)NAT * 4);
  float* maxb    = (float*)alloc((size_t)NAT * 4);
  float* sumb    = (float*)alloc((size_t)NAT * 4);
  float* cnt     = (float*)alloc((size_t)NRG * 4);
  bf16* xbf  = (bf16*)alloc((size_t)NAT * 64  * 2);
  bf16* abf  = (bf16*)alloc((size_t)NAT * CD  * 2);
  bf16* ebf  = (bf16*)alloc((size_t)EAT * 160 * 2);   // gathered edge inputs / bf16 h
  bf16* rbf  = (bf16*)alloc((size_t)NRG * CD  * 2);
  bf16* cbf  = (bf16*)alloc((size_t)NRG * 160 * 2);
  bf16* ebfR = (bf16*)alloc((size_t)ERG * 160 * 2);
  bf16* hbfR = (bf16*)alloc((size_t)NRG * CD  * 2);
  bf16* obf  = (bf16*)alloc((size_t)NB  * CD  * 2);
  bf16* hbfB = (bf16*)alloc((size_t)NB  * CD  * 2);
  bf16* wpool = (bf16*)alloc(1200000);                // weight bf16 pool
  bf16* wp = wpool;

  // ---- launch helpers -----------------------------------------------------
  auto FILL = [&](float* p, float v, size_t n){
    k_fill<<<nblk(n), 256, 0, stream>>>(p, v, n); };
  auto CVT = [&](const float* s, bf16* d, int M, int K, int Kp){
    k_cvt2d<<<nblk((size_t)M * Kp), 256, 0, stream>>>(s, d, M, K, Kp); };
  auto WCVT = [&](const float* s, int N, int K, int Kp)->bf16*{
    bf16* d = wp; wp += (size_t)N * Kp; CVT(s, d, N, K, Kp); return d; };
  auto GEMM = [&](const bf16* A, const bf16* B, const float* bias, float* Cout,
                  int M, int N, int K, int act){
    dim3 g((M / 16 + 7) / 8, N / 128);
    k_gemm<<<g, 256, 0, stream>>>(A, B, bias, Cout, M, N, K, act); };

  // ---- weights -> bf16 (zero-padded K) ------------------------------------
  bf16* bWlin1 = WCVT(FP(9),  CD, 64, 64);
  bf16* bWres  = WCVT(FP(11), CD, CD, CD);
  bf16* bWalin = WCVT(FP(13), CD, CD, CD);
  bf16* bWc0l1 = WCVT(FP(17), CD, CD + 16, 160);
  bf16* bWc0l2 = WCVT(FP(18), CD, CD, CD);
  bf16* bWg1   = WCVT(FP(20), CD, CD, CD);
  bf16* bWp    = WCVT(FP(24), CD, CD, CD);
  bf16* bWpih  = WCVT(FP(28), C3, CD, CD);
  bf16* bWphh  = WCVT(FP(29), C3, CD, CD);
  bf16* bWrg   = WCVT(FP(32), CD, CD + 18, 160);
  bf16* bWr0l1 = WCVT(FP(36), CD, CD + 1, 160);
  bf16* bWr0l2 = WCVT(FP(37), CD, CD, CD);
  bf16* bW0ih  = WCVT(FP(39), C3, CD, CD);
  bf16* bW0hh  = WCVT(FP(40), C3, CD, CD);
  bf16* bWr1   = WCVT(FP(43), CD, CD, CD);
  bf16* bW1ih  = WCVT(FP(47), C3, CD, CD);
  bf16* bW1hh  = WCVT(FP(48), C3, CD, CD);
  bf16* bWmol  = WCVT(FP(51), CD, CD, CD);
  bf16* bWmih  = WCVT(FP(55), C3, CD, CD);
  bf16* bWmhh  = WCVT(FP(56), C3, CD, CD);

  // ---- atom stage ---------------------------------------------------------
  CVT(x_in, xbf, NAT, 64, 64);
  GEMM(xbf, bWlin1, FP(10), x1, NAT, CD, 64, 1);          // x = leaky(lin1(x))
  CVT(x1, abf, NAT, CD, CD);
  GEMM(abf, bWres,  FP(12), res0, NAT, CD, CD, 1);        // res0
  GEMM(abf, bWalin, FP(14), tA,   NAT, CD, CD, 1);        // leaky(alin(x))

  // GATEConv c0
  k_gather<<<nblk((size_t)EAT*160),256,0,stream>>>(x1, eattr, 16, srcA, ebf, EAT, 160);
  GEMM(ebf, bWc0l1, nullptr, hE, EAT, CD, 160, 1);        // h = leaky(lin1(concat))
  k_rowdot<<<nblk(EAT),256,0,stream>>>(hE, FP(15), aj, EAT);
  k_rowdot<<<nblk(NAT),256,0,stream>>>(x1, FP(16), sD, NAT);
  FILL(maxb, -1e30f, NAT); FILL(sumb, 0.f, NAT);
  k_escore<<<nblk(EAT),256,0,stream>>>(aj, nullptr, sD, nullptr, dstA, alpha, maxb, EAT);
  k_eexp  <<<nblk(EAT),256,0,stream>>>(alpha, maxb, dstA, sumb, EAT);
  k_enorm <<<nblk(EAT),256,0,stream>>>(alpha, sumb, dstA, EAT);
  CVT(hE, ebf, EAT, CD, CD);                              // reuse ebf as bf16 h
  GEMM(ebf, bWc0l2, nullptr, hE, EAT, CD, CD, 0);         // hm (overwrites hE)
  FILL(mA, 0.f, (size_t)NAT * CD);
  k_scatter<<<nblk((size_t)EAT*CD),256,0,stream>>>(hE, nullptr, dstA, alpha, mA, EAT);
  k_finalize<<<nblk((size_t)NAT*CD),256,0,stream>>>(mA, FP(19), tA, nullptr, x1, NAT, 1, 1);

  // GATConv g1
  CVT(x1, abf, NAT, CD, CD);
  GEMM(abf, bWalin, FP(14), tA, NAT, CD, CD, 1);
  GEMM(abf, bWg1, nullptr, mA, NAT, CD, CD, 0);           // hs = hd
  k_rowdot<<<nblk(NAT),256,0,stream>>>(mA, FP(21), sS, NAT);
  k_rowdot<<<nblk(NAT),256,0,stream>>>(mA, FP(22), sD, NAT);
  FILL(maxb, -1e30f, NAT); FILL(sumb, 0.f, NAT);
  k_escore<<<nblk(EAT),256,0,stream>>>(nullptr, sS, sD, srcA, dstA, alpha, maxb, EAT);
  k_eexp  <<<nblk(EAT),256,0,stream>>>(alpha, maxb, dstA, sumb, EAT);
  k_enorm <<<nblk(EAT),256,0,stream>>>(alpha, sumb, dstA, EAT);
  FILL(aggA, 0.f, (size_t)NAT * CD);
  k_scatter<<<nblk((size_t)EAT*CD),256,0,stream>>>(mA, srcA, dstA, alpha, aggA, EAT);
  k_finalize<<<nblk((size_t)NAT*CD),256,0,stream>>>(aggA, FP(23), tA, res0, x1, NAT, 1, 1);

  // ---- atom -> reduced-graph pooling --------------------------------------
  FILL(cnt, 0.f, NRG); FILL(rg_base, 0.f, (size_t)NRG * CD);
  k_count<<<nblk(NAT),256,0,stream>>>(pool_tid, cnt, NAT);
  k_scatter<<<nblk((size_t)NAT*CD),256,0,stream>>>(x1, pool_sid, pool_tid, nullptr, rg_base, NAT);
  k_div<<<nblk((size_t)NRG*CD),256,0,stream>>>(rg_base, cnt, NRG);

  // pconv GATConv (src = atoms, dst = rg_base; identical both passes)
  CVT(x1, abf, NAT, CD, CD);
  GEMM(abf, bWp, nullptr, mA, NAT, CD, CD, 0);            // hs (atoms)
  CVT(rg_base, rbf, NRG, CD, CD);
  GEMM(rbf, bWp, nullptr, tR, NRG, CD, CD, 0);            // hd (rg)
  k_rowdot<<<nblk(NAT),256,0,stream>>>(mA, FP(25), sS, NAT);
  k_rowdot<<<nblk(NRG),256,0,stream>>>(tR, FP(26), sD, NRG);
  FILL(maxb, -1e30f, NRG); FILL(sumb, 0.f, NRG);
  k_escore<<<nblk(NAT),256,0,stream>>>(nullptr, sS, sD, pool_sid, pool_tid, alpha, maxb, NAT);
  k_eexp  <<<nblk(NAT),256,0,stream>>>(alpha, maxb, pool_tid, sumb, NAT);
  k_enorm <<<nblk(NAT),256,0,stream>>>(alpha, sumb, pool_tid, NAT);
  FILL(aggR, 0.f, (size_t)NRG * CD);
  k_scatter<<<nblk((size_t)NAT*CD),256,0,stream>>>(mA, pool_sid, pool_tid, alpha, aggR, NAT);
  k_finalize<<<nblk((size_t)NRG*CD),256,0,stream>>>(aggR, FP(27), nullptr, nullptr, hR, NRG, 1, 0);
  // pool GRU x2 (h fixed -> gi fixed)
  k_copy<<<nblk((size_t)NRG*CD),256,0,stream>>>(rg_base, rg_out, (size_t)NRG * CD);
  CVT(hR, hbfR, NRG, CD, CD);
  GEMM(hbfR, bWpih, FP(30), giR, NRG, C3, CD, 0);
  for (int pass = 0; pass < 2; ++pass) {
    CVT(rg_out, rbf, NRG, CD, CD);
    GEMM(rbf, bWphh, FP(31), ghR, NRG, C3, CD, 0);
    k_gru<<<nblk((size_t)NRG*CD),256,0,stream>>>(giR, ghR, rg_out, rg_out, NRG, 1);
  }

  // lin_rg on concat(rg_out, rg_x)
  k_gather<<<nblk((size_t)NRG*160),256,0,stream>>>(rg_out, rg_x, 18, nullptr, cbf, NRG, 160);
  GEMM(cbf, bWrg, FP(33), rg_out, NRG, CD, 160, 0);

  // r0 GATEConv on rg graph
  k_gather<<<nblk((size_t)ERG*160),256,0,stream>>>(rg_out, rg_ea, 1, srcR, ebfR, ERG, 160);
  GEMM(ebfR, bWr0l1, nullptr, hER, ERG, CD, 160, 1);
  k_rowdot<<<nblk(ERG),256,0,stream>>>(hER, FP(34), aj, ERG);
  k_rowdot<<<nblk(NRG),256,0,stream>>>(rg_out, FP(35), sD, NRG);
  FILL(maxb, -1e30f, NRG); FILL(sumb, 0.f, NRG);
  k_escore<<<nblk(ERG),256,0,stream>>>(aj, nullptr, sD, nullptr, dstR, alpha, maxb, ERG);
  k_eexp  <<<nblk(ERG),256,0,stream>>>(alpha, maxb, dstR, sumb, ERG);
  k_enorm <<<nblk(ERG),256,0,stream>>>(alpha, sumb, dstR, ERG);
  CVT(hER, ebfR, ERG, CD, CD);
  GEMM(ebfR, bWr0l2, nullptr, hER, ERG, CD, CD, 0);
  FILL(aggR, 0.f, (size_t)NRG * CD);
  k_scatter<<<nblk((size_t)ERG*CD),256,0,stream>>>(hER, nullptr, dstR, alpha, aggR, ERG);
  k_finalize<<<nblk((size_t)NRG*CD),256,0,stream>>>(aggR, FP(38), nullptr, nullptr, hR, NRG, 1, 0);
  CVT(hR, hbfR, NRG, CD, CD);
  GEMM(hbfR, bW0ih, FP(41), giR, NRG, C3, CD, 0);
  CVT(rg_out, rbf, NRG, CD, CD);
  GEMM(rbf, bW0hh, FP(42), ghR, NRG, C3, CD, 0);
  k_gru<<<nblk((size_t)NRG*CD),256,0,stream>>>(giR, ghR, rg_out, rg_out, NRG, 1);

  // r1 GATConv on rg graph
  CVT(rg_out, rbf, NRG, CD, CD);
  GEMM(rbf, bWr1, nullptr, tR, NRG, CD, CD, 0);           // hs = hd
  k_rowdot<<<nblk(NRG),256,0,stream>>>(tR, FP(44), sS, NRG);
  k_rowdot<<<nblk(NRG),256,0,stream>>>(tR, FP(45), sD, NRG);
  FILL(maxb, -1e30f, NRG); FILL(sumb, 0.f, NRG);
  k_escore<<<nblk(ERG),256,0,stream>>>(nullptr, sS, sD, srcR, dstR, alpha, maxb, ERG);
  k_eexp  <<<nblk(ERG),256,0,stream>>>(alpha, maxb, dstR, sumb, ERG);
  k_enorm <<<nblk(ERG),256,0,stream>>>(alpha, sumb, dstR, ERG);
  FILL(aggR, 0.f, (size_t)NRG * CD);
  k_scatter<<<nblk((size_t)ERG*CD),256,0,stream>>>(tR, srcR, dstR, alpha, aggR, ERG);
  k_finalize<<<nblk((size_t)NRG*CD),256,0,stream>>>(aggR, FP(46), nullptr, nullptr, hR, NRG, 1, 0);
  CVT(hR, hbfR, NRG, CD, CD);
  GEMM(hbfR, bW1ih, FP(49), giR, NRG, C3, CD, 0);
  CVT(rg_out, rbf, NRG, CD, CD);
  GEMM(rbf, bW1hh, FP(50), ghR, NRG, C3, CD, 0);
  k_gru<<<nblk((size_t)NRG*CD),256,0,stream>>>(giR, ghR, rg_out, rg_out, NRG, 1);

  // ---- molecule readout ---------------------------------------------------
  FILL(outM, 0.f, (size_t)NB * CD);
  k_scatter<<<nblk((size_t)NRG*CD),256,0,stream>>>(rg_out, nullptr, rg_batch, nullptr, outM, NRG);
  k_finalize<<<nblk((size_t)NB*CD),256,0,stream>>>(outM, nullptr, nullptr, nullptr, outM, NB, 0, 1);
  CVT(rg_out, rbf, NRG, CD, CD);
  GEMM(rbf, bWmol, nullptr, tR, NRG, CD, CD, 0);          // hs fixed
  k_rowdot<<<nblk(NRG),256,0,stream>>>(tR, FP(52), sS, NRG);
  for (int pass = 0; pass < 2; ++pass) {
    CVT(outM, obf, NB, CD, CD);
    GEMM(obf, bWmol, nullptr, hB, NB, CD, CD, 0);         // hd
    k_rowdot<<<nblk(NB),256,0,stream>>>(hB, FP(53), sD, NB);
    FILL(maxb, -1e30f, NB); FILL(sumb, 0.f, NB);
    k_escore<<<nblk(NRG),256,0,stream>>>(nullptr, sS, sD, nullptr, rg_batch, alpha, maxb, NRG);
    k_eexp  <<<nblk(NRG),256,0,stream>>>(alpha, maxb, rg_batch, sumb, NRG);
    k_enorm <<<nblk(NRG),256,0,stream>>>(alpha, sumb, rg_batch, NRG);
    FILL(aggB, 0.f, (size_t)NB * CD);
    k_scatter<<<nblk((size_t)NRG*CD),256,0,stream>>>(tR, nullptr, rg_batch, alpha, aggB, NRG);
    k_finalize<<<nblk((size_t)NB*CD),256,0,stream>>>(aggB, FP(54), nullptr, nullptr, hB, NB, 1, 0);
    CVT(hB, hbfB, NB, CD, CD);
    GEMM(hbfB, bWmih, FP(57), giB, NB, C3, CD, 0);
    GEMM(obf,  bWmhh, FP(58), ghB, NB, C3, CD, 0);
    k_gru<<<nblk((size_t)NB*CD),256,0,stream>>>(giB, ghB, outM, outM, NB, 1);
  }

  // ---- outputs: pred [NB], fp [NB,CD] -------------------------------------
  float* dout = (float*)d_out;
  k_pred<<<nblk(NB),256,0,stream>>>(outM, FP(59), FP(60), dout, NB);
  k_copy<<<nblk((size_t)NB*CD),256,0,stream>>>(outM, dout + NB, (size_t)NB * CD);
#undef FP
}
